// SAGE_78151224918249
// MI455X (gfx1250) — matrix operand order repaired and checked by reading, not compiled, not run on hardware
//
#include <hip/hip_runtime.h>
#include <math.h>
#include <stdint.h>

typedef __attribute__((ext_vector_type(16))) _Float16 v16h;
typedef __attribute__((ext_vector_type(8)))  _Float16 v8h;
typedef __attribute__((ext_vector_type(8)))  float    v8f;

#define NNODES 50000
#define HID 128

// ---------------- zero (dword granularity) ----------------
__global__ void k_zero(float* __restrict__ p, int n) {
  int i = blockIdx.x * blockDim.x + threadIdx.x;
  int stride = gridDim.x * blockDim.x;
  for (; i < n; i += stride) p[i] = 0.f;
}

// ---------------- degree (deg[dst] += 1, f32 atomics) ----------------
__global__ void k_degree(const int* __restrict__ dst, int E, float* __restrict__ deg) {
  int i = blockIdx.x * blockDim.x + threadIdx.x;
  int stride = gridDim.x * blockDim.x;
  for (; i < E; i += stride) unsafeAtomicAdd(&deg[dst[i]], 1.0f);
}

// ---------------- scatter-add: out[dst] += feat[src] (f16, packed atomics) ----------
// One wave per edge: lane owns 4 halfs (8 B gather) -> two global_atomic_pk_add_f16.
__global__ __launch_bounds__(256)
void k_scatter(const _Float16* __restrict__ feat, const int* __restrict__ src,
               const int* __restrict__ dst, _Float16* __restrict__ out, int E) {
  int gid  = blockIdx.x * blockDim.x + threadIdx.x;
  int lane = gid & 31;
  int warp = gid >> 5;
  int nw   = (gridDim.x * blockDim.x) >> 5;
  for (int e = warp; e < E; e += nw) {
    int s = src[e], d = dst[e];
    if (e + nw < E) {  // prefetch next edge's source row (global_prefetch_b8)
      __builtin_prefetch(feat + (size_t)src[e + nw] * HID + lane * 4, 0, 0);
    }
    const uint2 v = *(const uint2*)(feat + (size_t)s * HID + lane * 4);
    const _Float16* o = out + (size_t)d * HID + lane * 4;
    asm volatile("global_atomic_pk_add_f16 %0, %1, off"
                 :: "v"(o), "v"(v.x) : "memory");
    asm volatile("global_atomic_pk_add_f16 %0, %1, off offset:4"
                 :: "v"(o), "v"(v.y) : "memory");
  }
}

// ---------------- weight repack: f32 [Ktot x 128] -> WMMA B-fragment order (f16) ----
// B fragment (16x16x32 f16, per 32-K step): lane holds column n = 16c + (lane&15);
// VGPR v holds K pair k = s*32 + (lane>>4)*8 + 2v + (v>>2)*8. Packed so each lane's 8 dwords of
// a step are contiguous: dword index = ((c*ksteps + s)*32 + lane)*8 + v.
__global__ void k_repack(const float* __restrict__ Wtop, const float* __restrict__ Wbot,
                         int Ktop, int Ktot, _Float16* __restrict__ out) {
  int ksteps = Ktot >> 5;
  int total  = 8 * ksteps * 32 * 8;   // dwords
  int i = blockIdx.x * blockDim.x + threadIdx.x;
  int stride = gridDim.x * blockDim.x;
  for (; i < total; i += stride) {
    int v    = i & 7;
    int lane = (i >> 3) & 31;
    int cs   = i >> 8;
    int s    = cs % ksteps;
    int c    = cs / ksteps;
    int n = c * 16 + (lane & 15);
    int k = s * 32 + ((lane >> 4) << 3) + (v << 1) + ((v >> 2) << 3);
    float x0 = (k     < Ktop) ? Wtop[(size_t)k * 128 + n]       : Wbot[(size_t)(k - Ktop) * 128 + n];
    float x1 = (k + 1 < Ktop) ? Wtop[(size_t)(k + 1) * 128 + n] : Wbot[(size_t)(k + 1 - Ktop) * 128 + n];
    out[(size_t)i * 2 + 0] = (_Float16)x0;
    out[(size_t)i * 2 + 1] = (_Float16)x1;
  }
}

// ---------------- register-resident B fragments (one col tile per wave) -------------
template <int KSTEPS>
__device__ __forceinline__ void load_bfrags(const _Float16* __restrict__ Bpk,
                                            int wave, int lane, v16h (&bfr)[KSTEPS]) {
#pragma unroll
  for (int s = 0; s < KSTEPS; ++s) {
    const _Float16* bp = Bpk + ((size_t)(wave * KSTEPS + s) * 32 + lane) * 16;
    v8h blo = *(const v8h*)(bp);
    v8h bhi = *(const v8h*)(bp + 8);
    bfr[s] = __builtin_shufflevector(blo, bhi,
        0, 1, 2, 3, 4, 5, 6, 7, 8, 9, 10, 11, 12, 13, 14, 15);
  }
}

// ---------------- WMMA tile: 32 rows x 128 cols per block, 2 accumulators/wave ------
// A fragments stream from LDS (ds_load_b128), B lives in registers; the two WMMAs per
// step are independent, covering the D->A/B hazard window.
template <int KSTEPS, int ACT>
__device__ __forceinline__ void wmma_tile(const _Float16* As, const v16h (&bfr)[KSTEPS],
                                          const float* __restrict__ bias,
                                          _Float16* __restrict__ Cout,
                                          int r0, int Nrows, float scale, int tid) {
  constexpr int Ktot = KSTEPS * 32;
  const int wave = tid >> 5;
  const int lane = tid & 31;
  const int arow = (lane & 15) * Ktot + ((lane >> 4) << 3);

  v8f acc0 = {};
  v8f acc1 = {};
#pragma unroll
  for (int s = 0; s < KSTEPS; ++s) {
    const _Float16* ap0 = As + arow + s * 32;           // rows r0 .. r0+15
    const _Float16* ap1 = ap0 + 16 * Ktot;              // rows r0+16 .. r0+31
    v8h a0lo = *(const v8h*)(ap0);
    v8h a0hi = *(const v8h*)(ap0 + 16);
    v8h a1lo = *(const v8h*)(ap1);
    v8h a1hi = *(const v8h*)(ap1 + 16);
    v16h a0 = __builtin_shufflevector(a0lo, a0hi,
        0, 1, 2, 3, 4, 5, 6, 7, 8, 9, 10, 11, 12, 13, 14, 15);
    v16h a1 = __builtin_shufflevector(a1lo, a1hi,
        0, 1, 2, 3, 4, 5, 6, 7, 8, 9, 10, 11, 12, 13, 14, 15);
    acc0 = __builtin_amdgcn_wmma_f32_16x16x32_f16(false, a0, false, bfr[s],
                                                  (short)0, acc0, false, false);
    acc1 = __builtin_amdgcn_wmma_f32_16x16x32_f16(false, a1, false, bfr[s],
                                                  (short)0, acc1, false, false);
  }

  const int n  = wave * 16 + (lane & 15);
  const float bn = bias[n];
#pragma unroll
  for (int j = 0; j < 8; ++j) {
    int m = j + ((lane >> 4) << 3);
    int r = r0 + m;
    if (r < Nrows) {
      float v = acc0[j] + bn;
      if (ACT == 1) v = scale * cosf(v);
      Cout[(size_t)r * 128 + n] = (_Float16)v;
    }
    int r2 = r + 16;
    if (r2 < Nrows) {
      float v = acc1[j] + bn;
      if (ACT == 1) v = scale * cosf(v);
      Cout[(size_t)r2 * 128 + n] = (_Float16)v;
    }
  }
}

// ---------------- RBF lift: C = f16( scale * cos(X_f32 @ Bpk + bias) ) --------------
template <int KSTEPS>
__global__ __launch_bounds__(256)
void k_gemm_rbf(const float* __restrict__ X,
                const _Float16* __restrict__ Bpk,
                const float* __restrict__ bias,
                _Float16* __restrict__ Cout, int Nrows, float scale) {
  constexpr int K0 = KSTEPS * 32;
  __shared__ _Float16 As[32 * K0];           // <= 8 KB
  const int r0  = blockIdx.x * 32;
  const int tid = threadIdx.x;
  v16h bfr[KSTEPS];
  load_bfrags<KSTEPS>(Bpk, tid >> 5, tid & 31, bfr);
  for (int e = tid; e < 32 * K0; e += 256) {
    int row = e / K0;
    int k   = e - row * K0;
    int r   = r0 + row;
    float v = (r < Nrows) ? X[(size_t)r * K0 + k] : 0.f;
    As[row * K0 + k] = (_Float16)v;
  }
  __syncthreads();
  wmma_tile<KSTEPS, 1>(As, bfr, bias, Cout, r0, Nrows, scale, tid);
}

// ---------------- SAGE layer: C = f16( [feat | neigh/deg] @ Bpk + bias ) ------------
// Persistent workgroups, B register-resident, LDS double-buffered. Self half of each
// A tile staged via global_load_async_to_lds_b128 two tiles deep (wait asynccnt<=2);
// neighbor half needs the 1/deg scale -> VALU + ds stores. Out-of-range rows clamp
// their source address so every wave issues exactly 2 async ops per tile (keeps the
// in-order ASYNCcnt pipeline bookkeeping exact); their outputs are store-guarded.
__global__ __launch_bounds__(256)
void k_gemm_sage(const _Float16* __restrict__ feat,
                 const _Float16* __restrict__ neigh,
                 const float* __restrict__ deg,
                 const _Float16* __restrict__ Bpk,
                 const float* __restrict__ bias,
                 _Float16* __restrict__ Cout, int Nrows, int ntiles) {
  __shared__ _Float16 As[2][32 * 256];       // 2 x 16 KB, row = [self 128 | neigh 128]
  const int tid = threadIdx.x;
  v16h bfr[8];
  load_bfrags<8>(Bpk, tid >> 5, tid & 31, bfr);

  auto stage = [&](int t, int b) {
    const int r0 = t * 32;
    const unsigned ldsbase = (unsigned)(uintptr_t)(&As[b][0]);
    // self half: 32 rows x 256 B in 16-B chunks, async global -> LDS (2 ops/wave)
#pragma unroll
    for (int ii = 0; ii < 2; ++ii) {
      int idx  = tid + ii * 256;
      int row  = idx >> 4;                   // 16 chunks per row
      int coff = (idx & 15) << 4;            // byte offset in self half
      int r = r0 + row;
      if (r >= Nrows) r = Nrows - 1;         // clamp: always issue (outputs guarded)
      unsigned    loff = ldsbase + row * 512 + coff;   // LDS row stride = 512 B
      const char* g    = (const char*)(feat + (size_t)r * 128) + coff;
      asm volatile("global_load_async_to_lds_b128 %0, %1, off"
                   :: "v"(loff), "v"(g) : "memory");
    }
    // neighbor half: rows scaled by 1/deg (0 if deg==0)
#pragma unroll
    for (int ii = 0; ii < 16; ++ii) {
      int e   = tid + ii * 256;
      int row = e >> 7;
      int k   = e & 127;
      int r   = r0 + row;
      float val = 0.f;
      if (r < Nrows) {
        float d   = deg[r];
        float inv = (d > 0.f) ? (1.f / d) : 0.f;
        val = (float)neigh[(size_t)r * 128 + k] * inv;
      }
      As[b][row * 256 + 128 + k] = (_Float16)val;
    }
  };

  int t = blockIdx.x;
  int b = 0;
  if (t < ntiles) stage(t, 0);
  for (; t < ntiles; t += gridDim.x, b ^= 1) {
    int nxt = t + gridDim.x;
    if (nxt < ntiles) {
      stage(nxt, b ^ 1);                     // stage next tile into other buffer
      asm volatile("s_wait_asynccnt 2" ::: "memory");  // current tile's asyncs done
    } else {
      asm volatile("s_wait_asynccnt 0" ::: "memory");
    }
    __syncthreads();
    wmma_tile<8, 0>(&As[b][0], bfr, bias, Cout, t * 32, Nrows, 0.f, tid);
    __syncthreads();                         // buffer b free for re-staging
  }
}

// ---------------- column-mean pooling (f16 in, f32 sums; divide by N in head) -------
__global__ void k_pool(const _Float16* __restrict__ X, int Nrows, float* __restrict__ pooled) {
  int col   = threadIdx.x;                       // 0..127
  int chunk = (Nrows + gridDim.x - 1) / gridDim.x;
  int rbeg  = blockIdx.x * chunk;
  int rend  = rbeg + chunk; if (rend > Nrows) rend = Nrows;
  float s = 0.f;
  for (int r = rbeg; r < rend; ++r) s += (float)X[(size_t)r * 128 + col];
  unsafeAtomicAdd(&pooled[col], s);
}

// ---------------- MLP head on pooled [256] vector ----------------
__global__ __launch_bounds__(256)
void k_head(const float* __restrict__ pooled,
            const float* __restrict__ fc1_W, const float* __restrict__ fc1_b,
            const float* __restrict__ out_W, const float* __restrict__ out_b,
            const float* __restrict__ ro_W,  const float* __restrict__ ro_b,
            float* __restrict__ out, float invN) {
  __shared__ float y[256];
  __shared__ float h1[128];
  __shared__ float h2[64];
  int t = threadIdx.x;
  y[t] = pooled[t] * invN;
  __syncthreads();
  if (t < 128) {
    float a = fc1_b[t];
    for (int i = 0; i < 256; ++i) a += y[i] * fc1_W[i * 128 + t];
    h1[t] = fmaxf(a, 0.f);
  }
  __syncthreads();
  if (t < 64) {
    float a = out_b[t];
    for (int i = 0; i < 128; ++i) a += h1[i] * out_W[i * 64 + t];
    h2[t] = (a > 0.f) ? a : 0.01f * a;   // leaky_relu(0.01)
  }
  __syncthreads();
  if (t == 0) {
    float a = ro_b[0];
    for (int i = 0; i < 64; ++i) a += h2[i] * ro_W[i];
    out[0] = a;
  }
}

// =====================================================================================
extern "C" void kernel_launch(void* const* d_in, const int* in_sizes, int n_in,
                              void* d_out, int out_size, void* d_ws, size_t ws_size,
                              hipStream_t stream) {
  const float* feat0   = (const float*)d_in[0];
  const float* feat1   = (const float*)d_in[1];
  const int*   src0    = (const int*)d_in[2];
  const int*   dst0    = (const int*)d_in[3];
  const int*   src1    = (const int*)d_in[4];
  const int*   dst1    = (const int*)d_in[5];
  const float* rbf_W0  = (const float*)d_in[6];
  const float* rbf_b0  = (const float*)d_in[7];
  const float* rbf_W1  = (const float*)d_in[8];
  const float* rbf_b1  = (const float*)d_in[9];
  const float* Wself1  = (const float*)d_in[10];
  const float* Wneigh1 = (const float*)d_in[11];
  const float* b1      = (const float*)d_in[12];
  const float* Wself2  = (const float*)d_in[13];
  const float* Wneigh2 = (const float*)d_in[14];
  const float* b2      = (const float*)d_in[15];
  const float* fc1_W   = (const float*)d_in[16];
  const float* fc1_b   = (const float*)d_in[17];
  const float* out_W   = (const float*)d_in[18];
  const float* out_b   = (const float*)d_in[19];
  const float* ro_W    = (const float*)d_in[20];
  const float* ro_b    = (const float*)d_in[21];

  const int N = NNODES;
  const int E = in_sizes[2];

  // ---- workspace carve-out (256B aligned) ----
  char* ws = (char*)d_ws;
  auto carve = [&](size_t bytes) {
    char* p = ws;
    ws += (bytes + 255) & ~(size_t)255;
    return p;
  };
  _Float16* featA   = (_Float16*)carve((size_t)N * 128 * 2);
  _Float16* featB   = (_Float16*)carve((size_t)N * 128 * 2);
  _Float16* neigh   = (_Float16*)carve((size_t)N * 128 * 2);
  float*    deg     = (float*)carve((size_t)N * 4);
  float*    pooled  = (float*)carve(256 * 4);
  _Float16* pk_rbf0 = (_Float16*)carve(128 * 128 * 2);
  _Float16* pk_rbf1 = (_Float16*)carve(64 * 128 * 2);
  _Float16* pk_s1[3];
  _Float16* pk_s2[3];
  for (int l = 0; l < 3; ++l) pk_s1[l] = (_Float16*)carve(256 * 128 * 2);
  for (int l = 0; l < 3; ++l) pk_s2[l] = (_Float16*)carve(256 * 128 * 2);

  const int   nblk32   = (N + 31) / 32;              // 1563 row tiles
  const float rbfScale = sqrtf(2.0f / 128.0f);

  // ---- repack all GEMM weights into WMMA B-fragment order (f16) ----
  k_repack<<<32, 256, 0, stream>>>(rbf_W0, nullptr, 128, 128, pk_rbf0);
  k_repack<<<16, 256, 0, stream>>>(rbf_W1, nullptr, 64, 64, pk_rbf1);
  for (int l = 0; l < 3; ++l) {
    k_repack<<<64, 256, 0, stream>>>(Wself1 + (size_t)l * 128 * 128,
                                     Wneigh1 + (size_t)l * 128 * 128, 128, 256, pk_s1[l]);
    k_repack<<<64, 256, 0, stream>>>(Wself2 + (size_t)l * 128 * 128,
                                     Wneigh2 + (size_t)l * 128 * 128, 128, 256, pk_s2[l]);
  }

  k_zero<<<1, 256, 0, stream>>>(pooled, 256);

  auto run_branch = [&](const float* feat, int fdim, const _Float16* pkrbf,
                        const float* rbfb, _Float16* const* pks, const float* bias,
                        const int* src, const int* dst, float* pooldst) {
    // degrees (constant across layers)
    k_zero<<<196, 256, 0, stream>>>(deg, N);
    k_degree<<<3125, 256, 0, stream>>>(dst, E, deg);
    // RBF lift: featA = f16( sqrt(2/H) * cos(feat @ W + b) )
    if (fdim == 128) {
      k_gemm_rbf<4><<<nblk32, 256, 0, stream>>>(feat, pkrbf, rbfb, featA, N, rbfScale);
    } else {
      k_gemm_rbf<2><<<nblk32, 256, 0, stream>>>(feat, pkrbf, rbfb, featA, N, rbfScale);
    }
    // 3 SAGE layers
    for (int l = 0; l < 3; ++l) {
      k_zero<<<4096, 256, 0, stream>>>((float*)neigh, N * 64);   // N*128 halfs
      k_scatter<<<4096, 256, 0, stream>>>(featA, src, dst, neigh, E);
      k_gemm_sage<<<512, 256, 0, stream>>>(featA, neigh, deg, pks[l],
                                           bias + (size_t)l * 128, featB, N, nblk32);
      _Float16* t = featA; featA = featB; featB = t;
    }
    k_pool<<<256, 128, 0, stream>>>(featA, N, pooldst);
  };

  // y = concat(branch2 mean (g0 / feat0), branch1 mean (g1 / feat1))
  run_branch(feat0, 128, pk_rbf0, rbf_b0, pk_s2, b2, src0, dst0, pooled);
  run_branch(feat1, 64,  pk_rbf1, rbf_b1, pk_s1, b1, src1, dst1, pooled + 128);

  k_head<<<1, 256, 0, stream>>>(pooled, fc1_W, fc1_b, out_W, out_b, ro_W, ro_b,
                                (float*)d_out, 1.0f / (float)N);
}
